// ParticleRetention_25254407700896
// MI455X (gfx1250) — matrix operand order repaired and checked by reading, not compiled, not run on hardware
//
#include <hip/hip_runtime.h>
#include <hip/hip_bf16.h>

typedef __bf16 bf16;
typedef __attribute__((ext_vector_type(16))) __bf16 v16bf;
typedef __attribute__((ext_vector_type(8)))  float  v8f;

#define EMBED   512
#define T_SEQ   256
#define BATCH   128
#define HEADS   8
#define KD      64
#define N_ROWS  (BATCH * T_SEQ)     // 32768
#define SCALING 0.125f              // 64^-0.5

// ---------------------------------------------------------------------------
// gfx1250 async global->LDS helpers (ASYNCcnt-tracked, bypass VGPRs).
// LDS address operand = low 32 bits of the flat pointer (LDS aperture rule).
// ---------------------------------------------------------------------------
__device__ __forceinline__ void async_b128(void* lds_ptr, const void* gptr) {
    unsigned loff = (unsigned)(uintptr_t)lds_ptr;
    asm volatile("global_load_async_to_lds_b128 %0, %1, off"
                 :: "v"(loff), "v"(gptr) : "memory");
}
__device__ __forceinline__ void wait_async0() {
    asm volatile("s_wait_asynccnt 0x0" ::: "memory");
}

// ---------------------------------------------------------------------------
// One-time bf16 packing: x -> xb, {Wq,Wk,Wv,Wg,Wo} -> Wb[2560][512],
// biases -> bcat[2560].
// ---------------------------------------------------------------------------
__global__ __launch_bounds__(256) void cvt_x(const float* __restrict__ x,
                                             bf16* __restrict__ xb)
{
    size_t i = ((size_t)blockIdx.x * 256 + threadIdx.x) * 4;
    float4 v = *reinterpret_cast<const float4*>(x + i);
    xb[i + 0] = (bf16)v.x; xb[i + 1] = (bf16)v.y;
    xb[i + 2] = (bf16)v.z; xb[i + 3] = (bf16)v.w;
}

__global__ __launch_bounds__(256) void cvt_w(
    const float* __restrict__ W0, const float* __restrict__ W1,
    const float* __restrict__ W2, const float* __restrict__ W3,
    const float* __restrict__ W4,
    const float* __restrict__ b0, const float* __restrict__ b1,
    const float* __restrict__ b2, const float* __restrict__ b3,
    const float* __restrict__ b4,
    bf16* __restrict__ Wb, float* __restrict__ bcat)
{
    const int row = blockIdx.x;               // 0..2559
    const int mat = row >> 9;
    const int r   = row & 511;
    const float* W = mat == 0 ? W0 : mat == 1 ? W1 : mat == 2 ? W2 : mat == 3 ? W3 : W4;
    const float* b = mat == 0 ? b0 : mat == 1 ? b1 : mat == 2 ? b2 : mat == 3 ? b3 : b4;
    const int tid = threadIdx.x;
    Wb[(size_t)row * EMBED + tid]       = (bf16)W[(size_t)r * EMBED + tid];
    Wb[(size_t)row * EMBED + tid + 256] = (bf16)W[(size_t)r * EMBED + tid + 256];
    if (tid == 0) bcat[row] = b[r];
}

// ---------------------------------------------------------------------------
// Tiled bf16-WMMA GEMM  C[N,512] = A[N,512] @ W^T + bias, async-staged LDS
// with 2-deep double buffering. 256 threads = 8 waves; WG tile 128x64;
// wave tile 32x32 (2x2 WMMA 16x16x32 tiles).
// MODE 0: epilogue RoPE/scale/pack for q,k; bf16 pack for v; fp32 for g.
// MODE 1: fp32 output + bias (final projection).
// ---------------------------------------------------------------------------
constexpr int LDT = 40;                 // bf16 row stride: 80B = 5x16B aligned

__device__ __forceinline__ void stage_tiles(bf16* sA, bf16* sB,
                                            const bf16* Apt, const bf16* Wpt,
                                            int tid, int kb)
{
    // A tile 128x32: 16B chunks, 2 per thread
    #pragma unroll
    for (int u = 0; u < 2; ++u) {
        int cid = tid * 2 + u;
        int r = cid >> 2, c = cid & 3;
        async_b128(&sA[r * LDT + c * 8], Apt + (size_t)r * EMBED + kb + c * 8);
    }
    // B tile 64x32: 1 chunk per thread
    {
        int r = tid >> 2, c = tid & 3;
        async_b128(&sB[r * LDT + c * 8], Wpt + (size_t)r * EMBED + kb + c * 8);
    }
}

template <int MODE>
__global__ __launch_bounds__(256) void proj_gemm(
    const bf16* __restrict__ Asrc, const bf16* __restrict__ Wb,
    const float* __restrict__ bcat,
    const float* __restrict__ sinp, const float* __restrict__ cosp,
    bf16* __restrict__ qr, bf16* __restrict__ kr, bf16* __restrict__ vb,
    float* __restrict__ g, float* __restrict__ outp)
{
    __shared__ bf16 sA[2][128 * LDT];
    __shared__ bf16 sB[2][64 * LDT];

    const int tid  = threadIdx.x;
    const int lane = tid & 31;
    const int wave = tid >> 5;
    const int wr   = wave & 3;
    const int wc   = wave >> 2;
    const int lm   = lane & 15;
    const int lg   = lane >> 4;

    const int m0    = blockIdx.x * 128;
    const int mat   = (MODE == 0) ? (int)(blockIdx.y >> 3) : 4;
    const int cb    = (MODE == 0) ? (int)((blockIdx.y & 7) * 64) : (int)(blockIdx.y * 64);
    const int wrow0 = (MODE == 0) ? (int)(blockIdx.y * 64) : (int)(2048 + blockIdx.y * 64);

    const bf16* Apt = Asrc + (size_t)m0 * EMBED;
    const bf16* Wpt = Wb + (size_t)wrow0 * EMBED;

    v8f acc[2][2] = {};

    stage_tiles(sA[0], sB[0], Apt, Wpt, tid, 0);

    int buf = 0;
    for (int ks = 0; ks < 16; ++ks) {
        wait_async0();
        __syncthreads();                       // stage ks visible; prev buf free
        if (ks + 1 < 16)
            stage_tiles(sA[buf ^ 1], sB[buf ^ 1], Apt, Wpt, tid, (ks + 1) * 32);

        const bf16* cA = sA[buf];
        const bf16* cB = sB[buf];
        v16bf aF[2], bF[2];
        #pragma unroll
        for (int i = 0; i < 2; ++i) {
            const bf16* p = &cA[(wr * 32 + i * 16 + lm) * LDT + lg * 16];
            #pragma unroll
            for (int e = 0; e < 16; ++e) aF[i][e] = p[e];
        }
        #pragma unroll
        for (int j = 0; j < 2; ++j) {
            const bf16* p = &cB[(wc * 32 + j * 16 + lm) * LDT + lg * 16];
            #pragma unroll
            for (int e = 0; e < 16; ++e) bF[j][e] = p[e];
        }
        #pragma unroll
        for (int i = 0; i < 2; ++i)
            #pragma unroll
            for (int j = 0; j < 2; ++j)
                acc[i][j] = __builtin_amdgcn_wmma_f32_16x16x32_bf16(
                    false, aF[i], false, bF[j], (short)0, acc[i][j], false, false);
        buf ^= 1;
    }

    // Epilogue
    #pragma unroll
    for (int i = 0; i < 2; ++i) {
        #pragma unroll
        for (int j = 0; j < 2; ++j) {
            const int lc   = wc * 32 + j * 16 + lm;        // local col in [0,64)
            const int colg = cb + lc;                      // column in [0,512)
            const int h    = colg >> 6;
            const int d    = colg & 63;
            const float bia = bcat[wrow0 + lc];
            #pragma unroll
            for (int r = 0; r < 8; ++r) {
                const int n = m0 + wr * 32 + i * 16 + r + lg * 8;
                float val = acc[i][j][r] + bia;
                if constexpr (MODE == 1) {
                    outp[(size_t)n * EMBED + colg] = val;
                } else {
                    const int bidx = n >> 8;
                    const int t    = n & 255;
                    if (mat == 3) {
                        g[(size_t)n * EMBED + colg] = val;
                    } else if (mat == 2) {
                        vb[((size_t)(bidx * HEADS + h) * T_SEQ + t) * KD + d] = (bf16)val;
                    } else {
                        if (mat == 1) val *= SCALING;
                        // rotate_every_two partner: column d^1 == lane^1
                        float partner = __shfl_xor(val, 1, 32);
                        float rot = (d & 1) ? partner : -partner;
                        float res = val * cosp[t * KD + d] + rot * sinp[t * KD + d];
                        bf16* dst = (mat == 0) ? qr : kr;
                        dst[((size_t)(bidx * HEADS + h) * T_SEQ + t) * KD + d] = (bf16)res;
                    }
                }
            }
        }
    }
}

// ---------------------------------------------------------------------------
// Attention: one WG per (b,h, 64-query-row tile). All-in-LDS pipeline:
// S = Qr Kr^T + mask -> denom = max(|row sum|,1) -> P = S/denom (bf16)
// -> O = P V -> layernorm(64) -> silu(g)*O -> bf16. Aliased LDS phases.
// ---------------------------------------------------------------------------
__global__ __launch_bounds__(256) void attn_kernel(
    const bf16* __restrict__ qr, const bf16* __restrict__ kr,
    const bf16* __restrict__ vb, const float* __restrict__ g,
    const float* __restrict__ mask, bf16* __restrict__ attn)
{
    constexpr int LQK  = 72;   // bf16 stride sQ/sK (144B, 16B-aligned rows)
    constexpr int LVT  = 264;  // bf16 stride sVt/sSb
    constexpr int LS32 = 260;  // f32 stride sS
    constexpr int LO   = 68;   // f32 stride sO

    __shared__ __align__(16) unsigned char smem[146432];
    __shared__ float sRedA[256], sRedB[256], sDen[64], sMu[64], sRs[64];

    bf16*  sQ  = (bf16*)smem;                        // 64  x LQK
    bf16*  sK  = (bf16*)(smem + 9216);               // 256 x LQK
    bf16*  sVt = (bf16*)(smem + 46080);              // 64  x LVT
    float* sS  = (float*)(smem + 79872);             // 64  x LS32
    bf16*  sSb = (bf16*)smem;                        // alias over sQ+sK
    float* sO  = (float*)(smem + 79872);             // alias over sS

    const int tid  = threadIdx.x;
    const int lane = tid & 31;
    const int wave = tid >> 5;
    const int lm   = lane & 15;
    const int lg   = lane >> 4;
    const int rt   = blockIdx.x;
    const int bh   = blockIdx.y;
    const int bidx = bh >> 3;
    const int h    = bh & 7;
    const size_t base = (size_t)bh * T_SEQ * KD;

    // ---- Phase 1: async-stage Q(64x64), K(256x64); transpose V -> sVt ----
    {
        int row = tid >> 2, c0 = (tid & 3) * 16;
        const bf16* src = qr + base + (size_t)(rt * 64 + row) * KD + c0;
        async_b128(&sQ[row * LQK + c0], src);
        async_b128(&sQ[row * LQK + c0 + 8], src + 8);
    }
    {
        const bf16* src = kr + base + (size_t)tid * KD;
        #pragma unroll
        for (int e = 0; e < 8; ++e)
            async_b128(&sK[tid * LQK + e * 8], src + e * 8);
    }
    {
        int q4 = tid >> 6, d = tid & 63;
        for (int i = 0; i < 64; ++i) {
            int tt = q4 * 64 + i;
            sVt[d * LVT + tt] = vb[base + (size_t)tt * KD + d];
        }
    }
    wait_async0();
    __syncthreads();

    // ---- Phase 2: S = Q K^T (wave w owns key cols [w*32, w*32+32)) ----
    v8f acc[4][2] = {};
    #pragma unroll
    for (int ks = 0; ks < 2; ++ks) {
        const int kb = ks * 32;
        v16bf bF[2];
        #pragma unroll
        for (int j = 0; j < 2; ++j) {
            const bf16* p = &sK[(wave * 32 + j * 16 + lm) * LQK + kb + lg * 16];
            #pragma unroll
            for (int e = 0; e < 16; ++e) bF[j][e] = p[e];
        }
        #pragma unroll
        for (int i = 0; i < 4; ++i) {
            v16bf aF;
            const bf16* p = &sQ[(i * 16 + lm) * LQK + kb + lg * 16];
            #pragma unroll
            for (int e = 0; e < 16; ++e) aF[e] = p[e];
            #pragma unroll
            for (int j = 0; j < 2; ++j)
                acc[i][j] = __builtin_amdgcn_wmma_f32_16x16x32_bf16(
                    false, aF, false, bF[j], (short)0, acc[i][j], false, false);
        }
    }
    const float* mrow = mask + ((size_t)bh * T_SEQ + rt * 64) * T_SEQ;
    #pragma unroll
    for (int i = 0; i < 4; ++i)
        #pragma unroll
        for (int j = 0; j < 2; ++j) {
            const int tk = wave * 32 + j * 16 + lm;
            #pragma unroll
            for (int r = 0; r < 8; ++r) {
                const int tq = i * 16 + r + lg * 8;
                sS[tq * LS32 + tk] = acc[i][j][r] + mrow[(size_t)tq * T_SEQ + tk];
            }
        }
    __syncthreads();

    // ---- Phase 3: denom = max(|row sum|, 1) ----
    {
        int row = tid >> 2, seg = tid & 3;
        float s = 0.f;
        const float* p = &sS[row * LS32 + seg * 64];
        #pragma unroll
        for (int c = 0; c < 64; ++c) s += p[c];
        sRedA[row * 4 + seg] = s;
    }
    __syncthreads();
    if (tid < 64) {
        float tot = sRedA[tid * 4] + sRedA[tid * 4 + 1] + sRedA[tid * 4 + 2] + sRedA[tid * 4 + 3];
        sDen[tid] = 1.0f / fmaxf(fabsf(tot), 1.0f);
    }
    __syncthreads();

    // ---- Phase 4: P = S * (1/denom) as bf16 over dead Q/K region ----
    {
        int row = tid >> 2, seg = tid & 3;
        float den = sDen[row];
        const float* p = &sS[row * LS32 + seg * 64];
        bf16* q2 = &sSb[row * LVT + seg * 64];
        #pragma unroll
        for (int c = 0; c < 64; ++c) q2[c] = (bf16)(p[c] * den);
    }
    __syncthreads();

    // ---- Phase 5: O = P V (each wave: 1 row-tile x 2 col-tiles, K=256) ----
    const int m  = wave & 3;
    const int nb = (wave >> 2) * 2;
    v8f oacc[2] = {};
    for (int ks = 0; ks < 8; ++ks) {
        const int k0 = ks * 32;
        v16bf aF;
        const bf16* p = &sSb[(m * 16 + lm) * LVT + k0 + lg * 16];
        #pragma unroll
        for (int e = 0; e < 16; ++e) aF[e] = p[e];
        #pragma unroll
        for (int j = 0; j < 2; ++j) {
            v16bf bF;
            const bf16* q2 = &sVt[((nb + j) * 16 + lm) * LVT + k0 + lg * 16];
            #pragma unroll
            for (int e = 0; e < 16; ++e) bF[e] = q2[e];
            oacc[j] = __builtin_amdgcn_wmma_f32_16x16x32_bf16(
                false, aF, false, bF, (short)0, oacc[j], false, false);
        }
    }
    #pragma unroll
    for (int j = 0; j < 2; ++j)
        #pragma unroll
        for (int r = 0; r < 8; ++r)
            sO[(m * 16 + r + lg * 8) * LO + (nb + j) * 16 + lm] = oacc[j][r];
    __syncthreads();

    // ---- Phase 6: layernorm over head dim, silu(g) gate, bf16 out ----
    {
        int row = tid >> 2, seg = tid & 3;
        float s1 = 0.f, s2 = 0.f;
        const float* p = &sO[row * LO + seg * 16];
        #pragma unroll
        for (int c = 0; c < 16; ++c) { float v = p[c]; s1 += v; s2 += v * v; }
        sRedA[row * 4 + seg] = s1;
        sRedB[row * 4 + seg] = s2;
    }
    __syncthreads();
    if (tid < 64) {
        float s1 = sRedA[tid * 4] + sRedA[tid * 4 + 1] + sRedA[tid * 4 + 2] + sRedA[tid * 4 + 3];
        float s2 = sRedB[tid * 4] + sRedB[tid * 4 + 1] + sRedB[tid * 4 + 2] + sRedB[tid * 4 + 3];
        float mu  = s1 * (1.0f / 64.0f);
        float var = s2 * (1.0f / 64.0f) - mu * mu;
        sMu[tid] = mu;
        sRs[tid] = rsqrtf(var + 1e-6f);
    }
    __syncthreads();
    {
        int row = tid >> 2, seg = tid & 3;
        float mu = sMu[row], rs = sRs[row];
        const int n = bidx * T_SEQ + rt * 64 + row;
        const float* gp = g + (size_t)n * EMBED + h * KD + seg * 16;
        bf16* op = attn + (size_t)n * EMBED + h * KD + seg * 16;
        const float* p = &sO[row * LO + seg * 16];
        #pragma unroll
        for (int c = 0; c < 16; ++c) {
            float val = (p[c] - mu) * rs;
            float gv  = gp[c];
            float si  = gv / (1.0f + __expf(-gv));
            op[c] = (bf16)(val * si);
        }
    }
}

// ---------------------------------------------------------------------------
extern "C" void kernel_launch(void* const* d_in, const int* in_sizes, int n_in,
                              void* d_out, int out_size, void* d_ws, size_t ws_size,
                              hipStream_t stream)
{
    (void)in_sizes; (void)n_in; (void)out_size; (void)ws_size;

    const float* x    = (const float*)d_in[0];
    const float* sinp = (const float*)d_in[1];
    const float* cosp = (const float*)d_in[2];
    const float* mask = (const float*)d_in[3];
    const float* Wq   = (const float*)d_in[4];
    const float* bq   = (const float*)d_in[5];
    const float* Wk   = (const float*)d_in[6];
    const float* bk   = (const float*)d_in[7];
    const float* Wv   = (const float*)d_in[8];
    const float* bv   = (const float*)d_in[9];
    const float* Wg   = (const float*)d_in[10];
    const float* bg   = (const float*)d_in[11];
    const float* Wo   = (const float*)d_in[12];
    const float* bo   = (const float*)d_in[13];
    float* out = (float*)d_out;

    char* ws = (char*)d_ws;
    const size_t SZ_G = (size_t)N_ROWS * EMBED * sizeof(float);              // 64 MB
    const size_t SZ_Q = (size_t)BATCH * HEADS * T_SEQ * KD * sizeof(bf16);   // 32 MB
    const size_t SZ_X = (size_t)N_ROWS * EMBED * sizeof(bf16);               // 32 MB
    const size_t SZ_W = (size_t)2560 * EMBED * sizeof(bf16);

    float* g    = (float*)ws;
    bf16*  qr   = (bf16*)(ws + SZ_G);
    bf16*  kr   = (bf16*)(ws + SZ_G + SZ_Q);
    bf16*  vb   = (bf16*)(ws + SZ_G + 2 * SZ_Q);
    bf16*  attn = (bf16*)(ws + SZ_G + 3 * SZ_Q);
    bf16*  xb   = (bf16*)(ws + SZ_G + 4 * SZ_Q);
    bf16*  Wb   = (bf16*)(ws + SZ_G + 4 * SZ_Q + SZ_X);
    float* bcat = (float*)(ws + SZ_G + 4 * SZ_Q + SZ_X + SZ_W);

    // One-time bf16 packing of activations + all five weight matrices
    cvt_x<<<dim3(N_ROWS * EMBED / 1024), 256, 0, stream>>>(x, xb);
    cvt_w<<<dim3(2560), 256, 0, stream>>>(Wq, Wk, Wv, Wg, Wo,
                                          bq, bk, bv, bg, bo, Wb, bcat);

    // Fused QKVG projection (+RoPE epilogue): 32 col-tiles of 64 over 4 mats
    proj_gemm<0><<<dim3(N_ROWS / 128, 32), 256, 0, stream>>>(
        xb, Wb, bcat, sinp, cosp, qr, kr, vb, g, nullptr);

    // Attention: 4 query-row tiles x (b*h)
    attn_kernel<<<dim3(4, BATCH * HEADS), 256, 0, stream>>>(qr, kr, vb, g, mask, attn);

    // Output projection
    proj_gemm<1><<<dim3(N_ROWS / 128, EMBED / 64), 256, 0, stream>>>(
        attn, Wb, bcat, sinp, cosp, qr, kr, vb, g, out);
}
